// MetaRNNBase_55422257988014
// MI455X (gfx1250) — matrix operand order repaired and checked by reading, not compiled, not run on hardware
//
#include <hip/hip_runtime.h>
#include <hip/hip_bf16.h>

// ---------------- types ----------------
typedef __bf16 bf16_t;
typedef __attribute__((ext_vector_type(8)))  bf16_t v8bf;
typedef __attribute__((ext_vector_type(16))) bf16_t v16bf;
typedef __attribute__((ext_vector_type(8)))  float  v8f;

#define DEV __device__ __forceinline__

// ---------------- problem constants ----------------
static constexpr int Bc  = 32;
static constexpr int Tc  = 256;
static constexpr int Hc  = 1024;
static constexpr int HYc = 256;
static constexpr int Ec  = 64;
static constexpr int Lc  = 2;
static constexpr int G4H  = 4 * Hc;    // 4096
static constexpr int G4HY = 4 * HYc;   // 1024
static constexpr int G4E  = 4 * Ec;    // 256

// ---------------- WMMA tile helpers (bf16, 16x16x32, f32 acc) ----------------
// A fragment: 16x32 tile of A (row-major, leading dim lda), rows = lane&15.
// lanes 0-15 hold K {k0+0..7, k0+16..23}; lanes 16-31 hold K {k0+8..15, k0+24..31}.
DEV v16bf ldA(const bf16_t* __restrict__ A, long lda, int k0, int lane) {
  const bf16_t* p = A + (size_t)(lane & 15) * (size_t)lda
                      + (size_t)(k0 + ((lane >> 4) << 3));
  v8bf lo = *(const v8bf*)p;
  v8bf hi = *(const v8bf*)(p + 16);
  return __builtin_shufflevector(lo, hi, 0,1,2,3,4,5,6,7,8,9,10,11,12,13,14,15);
}
// B fragment for C = A @ W^T: B[k][n] = W[n][k]. W row-major (N x K), ldw.
// lanes 0-15: N=lane, K={k0+0..7, k0+16..23} ; lanes 16-31: N=lane-16, K={k0+8..15, k0+24..31}.
DEV v16bf ldBt(const bf16_t* __restrict__ W, long ldw, int k0, int lane) {
  const bf16_t* p = W + (size_t)(lane & 15) * (size_t)ldw
                      + (size_t)(k0 + ((lane >> 4) << 3));
  v8bf lo = *(const v8bf*)p;
  v8bf hi = *(const v8bf*)(p + 16);
  return __builtin_shufflevector(lo, hi, 0,1,2,3,4,5,6,7,8,9,10,11,12,13,14,15);
}

DEV v8f wmma_bf16(v16bf a, v16bf b, v8f c) {
  return __builtin_amdgcn_wmma_f32_16x16x32_bf16(false, a, false, b,
                                                 (short)0, c, false, false);
}

// acc[j] += A_tile(16xK) @ W_tile_j(16xK)^T for NT adjacent 16-col tiles.
// Ping-pong double buffering: while WMMAs consume one fragment set, the loads
// for the other set are in flight. Requires K % 64 == 0.
template <int NT>
DEV void gemm_acc_nt(v8f (&acc)[NT], const bf16_t* __restrict__ A, long lda,
                     const bf16_t* __restrict__ W, long ldw, int K, int lane) {
  v16bf a0, b0[NT], a1, b1[NT];
  a0 = ldA(A, lda, 0, lane);
#pragma unroll
  for (int j = 0; j < NT; ++j)
    b0[j] = ldBt(W + (size_t)(16 * j) * (size_t)ldw, ldw, 0, lane);
  for (int k0 = 0; k0 < K; k0 += 64) {
    // issue loads for chunk k0+32 before consuming chunk k0
    a1 = ldA(A, lda, k0 + 32, lane);
#pragma unroll
    for (int j = 0; j < NT; ++j)
      b1[j] = ldBt(W + (size_t)(16 * j) * (size_t)ldw, ldw, k0 + 32, lane);
#pragma unroll
    for (int j = 0; j < NT; ++j) acc[j] = wmma_bf16(a0, b0[j], acc[j]);
    if (k0 + 64 < K) {
      a0 = ldA(A, lda, k0 + 64, lane);
#pragma unroll
      for (int j = 0; j < NT; ++j)
        b0[j] = ldBt(W + (size_t)(16 * j) * (size_t)ldw, ldw, k0 + 64, lane);
    }
#pragma unroll
    for (int j = 0; j < NT; ++j) acc[j] = wmma_bf16(a1, b1[j], acc[j]);
  }
}

DEV float sigm(float x) { return 1.0f / (1.0f + __expf(-x)); }

// ---------------- setup kernels ----------------
__global__ __launch_bounds__(256) void k_cvt(const float* __restrict__ s,
                                             bf16_t* __restrict__ d, long n) {
  long i = (long)blockIdx.x * blockDim.x + threadIdx.x;
  if (i < n) d[i] = (bf16_t)s[i];
}
__global__ __launch_bounds__(256) void k_zero_f32(float* p, long n) {
  long i = (long)blockIdx.x * blockDim.x + threadIdx.x;
  if (i < n) p[i] = 0.0f;
}
__global__ __launch_bounds__(256) void k_zero_bf16(bf16_t* p, long n) {
  long i = (long)blockIdx.x * blockDim.x + threadIdx.x;
  if (i < n) p[i] = (bf16_t)0.0f;
}

// ---------------- hyper LSTM pre-activation ----------------
// hg(B x 4HY) = x @ Wih[:, :H]^T + h @ Wih[:, H:]^T + hh @ Whh^T + bias
__global__ __launch_bounds__(128) void k_hyper_pre(
    const bf16_t* __restrict__ xbf, long ldx,
    const bf16_t* __restrict__ hbf,
    const bf16_t* __restrict__ hhbf,
    const bf16_t* __restrict__ Wih,   // (4HY x 2H)
    const bf16_t* __restrict__ Whh,   // (4HY x HY)
    const float*  __restrict__ bias,  // (4HY)
    float* __restrict__ hg) {
  int lane = threadIdx.x & 31, wid = threadIdx.x >> 5;
  int n0 = (blockIdx.x * 4 + wid) * 64;
  int r0 = blockIdx.y * 16;
  v8f acc[4] = {};
  gemm_acc_nt<4>(acc, xbf  + (size_t)r0 * ldx, ldx,
                 Wih + (size_t)n0 * (2 * Hc),      2 * Hc, Hc,  lane);
  gemm_acc_nt<4>(acc, hbf  + (size_t)r0 * Hc,  Hc,
                 Wih + (size_t)n0 * (2 * Hc) + Hc, 2 * Hc, Hc,  lane);
  gemm_acc_nt<4>(acc, hhbf + (size_t)r0 * HYc, HYc,
                 Whh + (size_t)n0 * HYc,           HYc,    HYc, lane);
  int rb = r0 + ((lane >> 4) << 3);
#pragma unroll
  for (int j = 0; j < 4; ++j) {
    int n = n0 + 16 * j + (lane & 15);
    float bn = bias[n];
#pragma unroll
    for (int i = 0; i < 8; ++i)
      hg[(size_t)(rb + i) * G4HY + n] = acc[j][i] + bn;
  }
}

// hyper pointwise LSTM: updates hc (f32 carry) and hh (bf16 carry)
__global__ __launch_bounds__(256) void k_hyper_point(
    const float* __restrict__ hg, float* __restrict__ hc,
    bf16_t* __restrict__ hhbf) {
  int idx = blockIdx.x * blockDim.x + threadIdx.x;
  if (idx >= Bc * HYc) return;
  int b = idx / HYc, j = idx % HYc;
  const float* g = hg + (size_t)b * G4HY;
  float hi = g[j], hf = g[HYc + j], hgc = g[2 * HYc + j], ho = g[3 * HYc + j];
  float c2 = sigm(hf) * hc[idx] + sigm(hi) * tanhf(hgc);
  float h2 = sigm(ho) * tanhf(c2);
  hc[idx]   = c2;
  hhbf[idx] = (bf16_t)h2;
}

// ---------------- z projections: z = hh2 @ Wz^T (+b), output bf16 ----------------
__global__ __launch_bounds__(128) void k_z(
    const bf16_t* __restrict__ hhbf,
    const bf16_t* __restrict__ Wzh, const bf16_t* __restrict__ Wzx,
    const bf16_t* __restrict__ Wzb,
    const float* __restrict__ bzh, const float* __restrict__ bzx,
    bf16_t* __restrict__ zh, bf16_t* __restrict__ zx, bf16_t* __restrict__ zb) {
  int lane = threadIdx.x & 31, wid = threadIdx.x >> 5;
  int which = blockIdx.z;
  const bf16_t* W = (which == 0) ? Wzh : (which == 1) ? Wzx : Wzb;
  const float* bias = (which == 0) ? bzh : (which == 1) ? bzx : nullptr;
  bf16_t* z = (which == 0) ? zh : (which == 1) ? zx : zb;
  int n0 = wid * 64;            // 4 waves cover N=256
  int r0 = blockIdx.y * 16;
  v8f acc[4] = {};
  gemm_acc_nt<4>(acc, hhbf + (size_t)r0 * HYc, HYc,
                 W + (size_t)n0 * HYc, HYc, HYc, lane);
  int rb = r0 + ((lane >> 4) << 3);
#pragma unroll
  for (int j = 0; j < 4; ++j) {
    int n = n0 + 16 * j + (lane & 15);
    float bn = bias ? bias[n] : 0.0f;
#pragma unroll
    for (int i = 0; i < 8; ++i)
      z[(size_t)(rb + i) * G4E + n] = (bf16_t)(acc[j][i] + bn);
  }
}

// ---------------- d projections: d[b, g*H+n] = z[b, g*E:] @ Wd[g]^T ----------------
__global__ __launch_bounds__(128) void k_d(
    const bf16_t* __restrict__ zh, const bf16_t* __restrict__ zx,
    const bf16_t* __restrict__ zb,
    const bf16_t* __restrict__ Wdh, const bf16_t* __restrict__ Wdx,
    const bf16_t* __restrict__ Wdb,                      // each (4,H,E) flat
    const float* __restrict__ b0,                        // (4H), added to db
    float* __restrict__ dh, float* __restrict__ dx, float* __restrict__ db) {
  int lane = threadIdx.x & 31, wid = threadIdx.x >> 5;
  int which = blockIdx.z / 4, g = blockIdx.z % 4;
  const bf16_t* z = (which == 0) ? zh : (which == 1) ? zx : zb;
  const bf16_t* W = (which == 0) ? Wdh : (which == 1) ? Wdx : Wdb;
  float* dst = (which == 0) ? dh : (which == 1) ? dx : db;
  int n0 = (blockIdx.x * 4 + wid) * 64;
  int r0 = blockIdx.y * 16;
  v8f acc[4] = {};
  gemm_acc_nt<4>(acc, z + (size_t)r0 * G4E + g * Ec, G4E,
                 W + ((size_t)g * Hc + n0) * Ec, Ec, Ec, lane);
  int rb = r0 + ((lane >> 4) << 3);
#pragma unroll
  for (int j = 0; j < 4; ++j) {
    int n = n0 + 16 * j + (lane & 15);
    float bn = (which == 2) ? b0[g * Hc + n] : 0.0f;
#pragma unroll
    for (int i = 0; i < 8; ++i)
      dst[(size_t)(rb + i) * G4H + g * Hc + n] = acc[j][i] + bn;
  }
}

// ---------------- main gates: pre = dh*(h@Whh^T) + dx*(x@Wih^T) + db ----------------
// Two sequential double-buffered GEMM loops (h then x); accumulators stay live.
__global__ __launch_bounds__(128) void k_main_pre(
    const bf16_t* __restrict__ hbf,
    const bf16_t* __restrict__ xbf, long ldx,
    const bf16_t* __restrict__ Whh, const bf16_t* __restrict__ Wih, // (4H x H)
    const float* __restrict__ dh, const float* __restrict__ dx,
    const float* __restrict__ db,
    float* __restrict__ pre) {
  int lane = threadIdx.x & 31, wid = threadIdx.x >> 5;
  int n0 = (blockIdx.x * 4 + wid) * 64;
  int r0 = blockIdx.y * 16;
  v8f ah[4] = {}, ax[4] = {};
  gemm_acc_nt<4>(ah, hbf + (size_t)r0 * Hc,  Hc,
                 Whh + (size_t)n0 * Hc, Hc, Hc, lane);
  gemm_acc_nt<4>(ax, xbf + (size_t)r0 * ldx, ldx,
                 Wih + (size_t)n0 * Hc, Hc, Hc, lane);
  int rb = r0 + ((lane >> 4) << 3);
#pragma unroll
  for (int j = 0; j < 4; ++j) {
    int n = n0 + 16 * j + (lane & 15);
#pragma unroll
    for (int i = 0; i < 8; ++i) {
      size_t o = (size_t)(rb + i) * G4H + n;
      pre[o] = dh[o] * ah[j][i] + dx[o] * ax[j][i] + db[o];
    }
  }
}

// main pointwise LSTM: updates c (f32), h (bf16 carry), optional outputs
__global__ __launch_bounds__(256) void k_main_point(
    const float* __restrict__ pre, float* __restrict__ c,
    bf16_t* __restrict__ hbf,
    float* __restrict__ y, long ystride,       // nullable: h_seq slice
    float* __restrict__ hlast, float* __restrict__ clast) {  // nullable
  int idx = blockIdx.x * blockDim.x + threadIdx.x;
  if (idx >= Bc * Hc) return;
  int b = idx / Hc, j = idx % Hc;
  const float* p = pre + (size_t)b * G4H;
  float gi = p[j], gf = p[Hc + j], gg = p[2 * Hc + j], go = p[3 * Hc + j];
  float c2 = sigm(gf) * c[idx] + sigm(gi) * tanhf(gg);
  float h2 = sigm(go) * tanhf(c2);
  c[idx]   = c2;
  hbf[idx] = (bf16_t)h2;
  if (y)     y[(size_t)b * ystride + j] = h2;
  if (hlast) { hlast[idx] = h2; clast[idx] = c2; }
}

// ---------------- host driver ----------------
extern "C" void kernel_launch(void* const* d_in, const int* in_sizes, int n_in,
                              void* d_out, int out_size, void* d_ws, size_t ws_size,
                              hipStream_t stream) {
  (void)in_sizes; (void)n_in; (void)out_size; (void)ws_size;
  const float* in    = (const float*)d_in[0];
  const float* w_ih  = (const float*)d_in[1];
  const float* w_hh  = (const float*)d_in[2];
  const float* hw_ih = (const float*)d_in[3];
  const float* hw_hh = (const float*)d_in[4];
  const float* hb    = (const float*)d_in[5];
  const float* w_zh  = (const float*)d_in[6];
  const float* b_zh  = (const float*)d_in[7];
  const float* w_zx  = (const float*)d_in[8];
  const float* b_zx  = (const float*)d_in[9];
  const float* w_zb  = (const float*)d_in[10];
  const float* w_dh  = (const float*)d_in[11];
  const float* w_dx  = (const float*)d_in[12];
  const float* w_db  = (const float*)d_in[13];
  const float* b0    = (const float*)d_in[14];
  float* out = (float*)d_out;

  // workspace carve-up (256B aligned)
  char* ws = (char*)d_ws;
  size_t off = 0;
  auto alloc = [&](size_t bytes) -> char* {
    char* p = ws + off;
    off = (off + bytes + 255) & ~(size_t)255;
    return p;
  };
  const long nIn   = (long)Bc * Tc * Hc;
  const long nWih  = (long)Lc * G4H * Hc;
  const long nWhh  = nWih;
  const long nHWih = (long)Lc * G4HY * 2 * Hc;
  const long nHWhh = (long)Lc * G4HY * HYc;
  const long nWz   = (long)Lc * G4E * HYc;
  const long nWd   = (long)Lc * 4 * Hc * Ec;

  bf16_t* in_bf   = (bf16_t*)alloc(nIn   * 2);
  bf16_t* wih_bf  = (bf16_t*)alloc(nWih  * 2);
  bf16_t* whh_bf  = (bf16_t*)alloc(nWhh  * 2);
  bf16_t* hwih_bf = (bf16_t*)alloc(nHWih * 2);
  bf16_t* hwhh_bf = (bf16_t*)alloc(nHWhh * 2);
  bf16_t* wzh_bf  = (bf16_t*)alloc(nWz * 2);
  bf16_t* wzx_bf  = (bf16_t*)alloc(nWz * 2);
  bf16_t* wzb_bf  = (bf16_t*)alloc(nWz * 2);
  bf16_t* wdh_bf  = (bf16_t*)alloc(nWd * 2);
  bf16_t* wdx_bf  = (bf16_t*)alloc(nWd * 2);
  bf16_t* wdb_bf  = (bf16_t*)alloc(nWd * 2);
  bf16_t* hbf_ws  = (bf16_t*)alloc((size_t)Lc * Bc * Hc  * 2);
  bf16_t* hhbf_ws = (bf16_t*)alloc((size_t)Lc * Bc * HYc * 2);
  float*  c_ws    = (float*) alloc((size_t)Lc * Bc * Hc  * 4);
  float*  hc_ws   = (float*) alloc((size_t)Lc * Bc * HYc * 4);
  float*  hg_ws   = (float*) alloc((size_t)Bc * G4HY * 4);
  bf16_t* zh_ws   = (bf16_t*)alloc((size_t)Bc * G4E * 2);
  bf16_t* zx_ws   = (bf16_t*)alloc((size_t)Bc * G4E * 2);
  bf16_t* zb_ws   = (bf16_t*)alloc((size_t)Bc * G4E * 2);
  float*  dh_ws   = (float*) alloc((size_t)Bc * G4H * 4);
  float*  dx_ws   = (float*) alloc((size_t)Bc * G4H * 4);
  float*  db_ws   = (float*) alloc((size_t)Bc * G4H * 4);
  float*  pre_ws  = (float*) alloc((size_t)Bc * G4H * 4);

  auto cvt = [&](const float* s, bf16_t* d, long n) {
    k_cvt<<<dim3((unsigned)((n + 255) / 256)), 256, 0, stream>>>(s, d, n);
  };
  cvt(in,    in_bf,   nIn);
  cvt(w_ih,  wih_bf,  nWih);
  cvt(w_hh,  whh_bf,  nWhh);
  cvt(hw_ih, hwih_bf, nHWih);
  cvt(hw_hh, hwhh_bf, nHWhh);
  cvt(w_zh,  wzh_bf,  nWz);
  cvt(w_zx,  wzx_bf,  nWz);
  cvt(w_zb,  wzb_bf,  nWz);
  cvt(w_dh,  wdh_bf,  nWd);
  cvt(w_dx,  wdx_bf,  nWd);
  cvt(w_db,  wdb_bf,  nWd);

  // zero carries
  {
    long n1 = (long)Lc * Bc * Hc, n2 = (long)Lc * Bc * HYc;
    k_zero_f32 <<<dim3((unsigned)((n1 + 255) / 256)), 256, 0, stream>>>(c_ws,  n1);
    k_zero_f32 <<<dim3((unsigned)((n2 + 255) / 256)), 256, 0, stream>>>(hc_ws, n2);
    k_zero_bf16<<<dim3((unsigned)((n1 + 255) / 256)), 256, 0, stream>>>(hbf_ws,  n1);
    k_zero_bf16<<<dim3((unsigned)((n2 + 255) / 256)), 256, 0, stream>>>(hhbf_ws, n2);
  }

  const long ldInput = (long)Tc * Hc;
  for (int t = 0; t < Tc; ++t) {
    for (int l = 0; l < Lc; ++l) {
      const bf16_t* xbf;
      long ldx;
      if (l == 0) { xbf = in_bf + (size_t)t * Hc; ldx = ldInput; }
      else        { xbf = hbf_ws;                 ldx = Hc; }      // layer-0 fresh h
      bf16_t* hbf_l  = hbf_ws  + (size_t)l * Bc * Hc;
      bf16_t* hhbf_l = hhbf_ws + (size_t)l * Bc * HYc;
      float*  c_l    = c_ws    + (size_t)l * Bc * Hc;
      float*  hc_l   = hc_ws   + (size_t)l * Bc * HYc;

      // 4 waves/block, each wave computes 16x64; block covers 256 cols.
      k_hyper_pre<<<dim3(G4HY / 256, Bc / 16), 128, 0, stream>>>(
          xbf, ldx, hbf_l, hhbf_l,
          hwih_bf + (size_t)l * G4HY * 2 * Hc,
          hwhh_bf + (size_t)l * G4HY * HYc,
          hb + (size_t)l * G4HY, hg_ws);

      k_hyper_point<<<dim3((Bc * HYc + 255) / 256), 256, 0, stream>>>(
          hg_ws, hc_l, hhbf_l);

      k_z<<<dim3(1, Bc / 16, 3), 128, 0, stream>>>(
          hhbf_l,
          wzh_bf + (size_t)l * G4E * HYc,
          wzx_bf + (size_t)l * G4E * HYc,
          wzb_bf + (size_t)l * G4E * HYc,
          b_zh + (size_t)l * G4E, b_zx + (size_t)l * G4E,
          zh_ws, zx_ws, zb_ws);

      k_d<<<dim3(Hc / 256, Bc / 16, 12), 128, 0, stream>>>(
          zh_ws, zx_ws, zb_ws,
          wdh_bf + (size_t)l * 4 * Hc * Ec,
          wdx_bf + (size_t)l * 4 * Hc * Ec,
          wdb_bf + (size_t)l * 4 * Hc * Ec,
          b0 + (size_t)l * G4H, dh_ws, dx_ws, db_ws);

      k_main_pre<<<dim3(G4H / 256, Bc / 16), 128, 0, stream>>>(
          hbf_l, xbf, ldx,
          whh_bf + (size_t)l * G4H * Hc,
          wih_bf + (size_t)l * G4H * Hc,
          dh_ws, dx_ws, db_ws, pre_ws);

      bool top = (l == Lc - 1);
      float* y  = top ? out + (size_t)t * Hc : nullptr;
      float* hl = (top && t == Tc - 1) ? out + (size_t)Bc * Tc * Hc : nullptr;
      float* cl = hl ? hl + (size_t)Bc * Hc : nullptr;
      k_main_point<<<dim3((Bc * Hc + 255) / 256), 256, 0, stream>>>(
          pre_ws, c_l, hbf_l, y, ldInput, hl, cl);
    }
  }
}
